// DeepSpeedEncoder_32830730011091
// MI455X (gfx1250) — matrix operand order repaired and checked by reading, not compiled, not run on hardware
//
#include <hip/hip_runtime.h>
#include <math.h>

typedef __bf16 bf16;
typedef __attribute__((ext_vector_type(16))) bf16  v16bf;
typedef __attribute__((ext_vector_type(8)))  bf16  v8bf;
typedef __attribute__((ext_vector_type(8)))  float v8f;
typedef int b128_t __attribute__((vector_size(16)));   // matches builtin param type

#define WMMA_BF16(a, b, c) \
  __builtin_amdgcn_wmma_f32_16x16x32_bf16(false, (a), false, (b), (short)0, (c), false, false)

// ---- CDNA5 async Global->LDS copy (ASYNCcnt path), with safe fallback ------
#if defined(__has_builtin)
#if __has_builtin(__builtin_amdgcn_global_load_async_to_lds_b128)
#define HAVE_ASYNC_LDS 1
#endif
#endif
#ifndef HAVE_ASYNC_LDS
#define HAVE_ASYNC_LDS 0
#endif

__device__ __forceinline__ void copy16_g2l(const bf16* g, bf16* l) {
#if HAVE_ASYNC_LDS
  __builtin_amdgcn_global_load_async_to_lds_b128(
      (__attribute__((address_space(1))) b128_t*)(g),
      (__attribute__((address_space(3))) b128_t*)(l), 0, 0);
#else
  *(uint4*)l = *(const uint4*)g;
#endif
}

__device__ __forceinline__ void wait_async_copies() {
#if HAVE_ASYNC_LDS
#if defined(__has_builtin) && __has_builtin(__builtin_amdgcn_s_wait_asynccnt)
  __builtin_amdgcn_s_wait_asynccnt(0);
#else
  asm volatile("s_wait_asynccnt 0" ::: "memory");
#endif
#endif
}

static constexpr int  BB    = 4;
static constexpr int  S     = 2048;
static constexpr int  H     = 1024;
static constexpr int  HEADS = 16;
static constexpr int  HD    = 64;       // head dim
static constexpr int  DFF   = 4096;
static constexpr long T     = (long)BB * S;   // 8192 tokens

// ---------------------------------------------------------------------------
// fp32 -> bf16 conversion (weights)
// ---------------------------------------------------------------------------
__global__ void cvt_bf16_kernel(const float* __restrict__ src, bf16* __restrict__ dst, long n) {
  long i      = (long)blockIdx.x * blockDim.x + threadIdx.x;
  long stride = (long)gridDim.x * blockDim.x;
  for (; i < n; i += stride) dst[i] = (bf16)src[i];
}

// ---------------------------------------------------------------------------
// LayerNorm over H=1024, one block (256 thr) per token, bf16 output
// ---------------------------------------------------------------------------
__global__ void ln_bf16_kernel(const float* __restrict__ x, const float* __restrict__ w,
                               const float* __restrict__ b, bf16* __restrict__ out) {
  __shared__ float red0[8], red1[8];
  const long t = blockIdx.x;
  const float* xr = x + t * H;
  float vals[4], s1 = 0.f, s2 = 0.f;
#pragma unroll
  for (int i = 0; i < 4; ++i) {
    float v = xr[threadIdx.x + i * 256];
    vals[i] = v; s1 += v; s2 += v * v;
  }
#pragma unroll
  for (int m = 16; m >= 1; m >>= 1) {
    s1 += __shfl_xor(s1, m, 32);
    s2 += __shfl_xor(s2, m, 32);
  }
  if ((threadIdx.x & 31) == 0) { red0[threadIdx.x >> 5] = s1; red1[threadIdx.x >> 5] = s2; }
  __syncthreads();
  s1 = 0.f; s2 = 0.f;
#pragma unroll
  for (int i = 0; i < 8; ++i) { s1 += red0[i]; s2 += red1[i]; }
  const float mu  = s1 * (1.f / (float)H);
  const float var = s2 * (1.f / (float)H) - mu * mu;
  const float inv = rsqrtf(var + 1e-12f);
#pragma unroll
  for (int i = 0; i < 4; ++i) {
    int c = threadIdx.x + i * 256;
    out[t * H + c] = (bf16)((vals[i] - mu) * inv * w[c] + b[c]);
  }
}

// ---------------------------------------------------------------------------
// GEMM core: C[128x128] += A[128xK] * B[KxN] (bf16 in, f32 acc), BK = 32
// 256 threads = 8 waves; wave (wm,wn) owns 64x32 => acc[4][2] of 16x16 tiles
// Double-buffered LDS staging; copies go through the CDNA5 async-LDS engine
// (ASYNCcnt) when available. One barrier + one s_wait_asynccnt per K-step.
// ---------------------------------------------------------------------------
__device__ __forceinline__ void gemm_core(const bf16* __restrict__ A, const bf16* __restrict__ B,
                                          int K, int ldb, v8f acc[4][2]) {
  __shared__ bf16 sA[2][128][40];    // 128 x 32 (+8 pad), x2 buffers
  __shared__ bf16 sB[2][32][136];    // 32 x 128 (+8 pad), x2 buffers
  const int tid  = threadIdx.x;
  const int lane = tid & 31, wave = tid >> 5;
  const int wm = wave & 1, wn = wave >> 1;
  const int half = lane >> 4, l16 = lane & 15;
  const long rowA0 = (long)blockIdx.y * 128;
  const int  colB0 = blockIdx.x * 128;

  v8f zero = {};
#pragma unroll
  for (int mi = 0; mi < 4; ++mi)
#pragma unroll
    for (int ni = 0; ni < 2; ++ni) acc[mi][ni] = zero;

  const int arow = tid >> 1, acol = (tid & 1) * 16;  // A tile: 128 rows x 32
  const int brow = tid >> 3, bcol = (tid & 7) * 16;  // B tile: 32 rows x 128

  const bf16* aBase = A + (rowA0 + arow) * (long)K + acol;   // + kb*32
  const bf16* bBase = B + (long)brow * ldb + colB0 + bcol;   // + kb*32*ldb

  auto copy_tile = [&](int buf, int kb) {
    const bf16* ap = aBase + kb * 32;
    const bf16* bp = bBase + (long)kb * 32 * ldb;
    copy16_g2l(ap,     &sA[buf][arow][acol]);
    copy16_g2l(ap + 8, &sA[buf][arow][acol + 8]);
    copy16_g2l(bp,     &sB[buf][brow][bcol]);
    copy16_g2l(bp + 8, &sB[buf][brow][bcol + 8]);
  };

  const int nk = K >> 5;
  copy_tile(0, 0);
  for (int kb = 0; kb < nk; ++kb) {
    const int cur = kb & 1, nxt = cur ^ 1;
    wait_async_copies();
    __syncthreads();                       // buffer `cur` ready; `nxt` free
    if (kb + 1 < nk) {
      copy_tile(nxt, kb + 1);              // overlap copy with WMMA below
      if (kb + 2 < nk) {                   // near-scope prefetch of kb+2
        __builtin_prefetch(aBase + (kb + 2) * 32, 0, 3);
        __builtin_prefetch(bBase + (long)(kb + 2) * 32 * ldb, 0, 3);
      }
    }
#pragma unroll
    for (int mi = 0; mi < 4; ++mi) {
      const int row = wm * 64 + mi * 16 + l16;
      v8bf alo = *(const v8bf*)&sA[cur][row][half * 8];        // K = 8*half .. +7
      v8bf ahi = *(const v8bf*)&sA[cur][row][16 + half * 8];   // K = 16+8*half .. +7
      v16bf a  = __builtin_shufflevector(alo, ahi, 0,1,2,3,4,5,6,7,8,9,10,11,12,13,14,15);
#pragma unroll
      for (int ni = 0; ni < 2; ++ni) {
        v16bf b;  // lane -> N column ; elems -> K (table order)
#pragma unroll
        for (int s = 0; s < 16; ++s)
          b[s] = sB[cur][s < 8 ? 8 * half + s : 16 + 8 * half + (s - 8)][wn * 32 + ni * 16 + l16];
        acc[mi][ni] = WMMA_BF16(a, b, acc[mi][ni]);
      }
    }
  }
}

#define GEMM_EPILOG_SETUP()                                        \
  const int tid = threadIdx.x;                                     \
  const int lane = tid & 31, wave = tid >> 5;                      \
  const int wm = wave & 1, wn = wave >> 1;                         \
  const int half = lane >> 4, l16 = lane & 15;                     \
  const long mbase = (long)blockIdx.y * 128 + wm * 64;             \
  const int  nbase = blockIdx.x * 128 + wn * 32;

// ---------------------------------------------------------------------------
// QKV GEMM: x_ln[T,H] @ Wqkv[H,3H] + b; scatter to q[b,h,s,d], kT[b,h,d,s],
// v[b,h,s,d]  (all bf16)
// ---------------------------------------------------------------------------
__global__ void gemm_qkv_kernel(const bf16* __restrict__ A, const bf16* __restrict__ B,
                                const float* __restrict__ bias,
                                bf16* __restrict__ qb, bf16* __restrict__ ktb,
                                bf16* __restrict__ vb) {
  v8f acc[4][2];
  gemm_core(A, B, H, 3 * H, acc);
  GEMM_EPILOG_SETUP();
#pragma unroll
  for (int mi = 0; mi < 4; ++mi)
#pragma unroll
    for (int ni = 0; ni < 2; ++ni) {
      const int n    = nbase + ni * 16 + l16;
      const int sect = n >> 10;              // 0=q 1=k 2=v
      const int hn   = n & 1023;
      const int head = hn >> 6, d = hn & 63;
      const float bv = bias[n];
#pragma unroll
      for (int r = 0; r < 8; ++r) {
        const long m = mbase + mi * 16 + half * 8 + r;  // token
        const long b = m >> 11;                         // /S
        const long s = m & 2047;
        const long bh = b * HEADS + head;
        const float val = acc[mi][ni][r] + bv;
        if (sect == 0)      qb [(bh * S + s) * HD + d]  = (bf16)val;
        else if (sect == 1) ktb[(bh * HD + d) * S + s]  = (bf16)val;   // transposed
        else                vb [(bh * S + s) * HD + d]  = (bf16)val;
      }
    }
}

// ---------------------------------------------------------------------------
// Flash attention: grid (S/64, B*HEADS), 128 threads (4 waves);
// each wave owns 16 q-rows; keys streamed in blocks of 64.
// ---------------------------------------------------------------------------
__global__ void attn_kernel(const bf16* __restrict__ q, const bf16* __restrict__ kT,
                            const bf16* __restrict__ v, const float* __restrict__ mask,
                            bf16* __restrict__ ctx) {
  const int lane = threadIdx.x & 31, wave = threadIdx.x >> 5;
  const int half = lane >> 4, l16 = lane & 15;
  const int bh = blockIdx.y;
  const int b  = bh >> 4, head = bh & 15;
  const int q0 = blockIdx.x * 64 + wave * 16;

  const bf16*  qp  = q  + ((long)bh * S + q0) * HD;   // [s][d]
  const bf16*  ktp = kT + (long)bh * HD * S;          // [d][s]
  const bf16*  vp  = v  + (long)bh * S * HD;          // [s][d]
  const float* mk  = mask + (long)b * S;

  __shared__ bf16 sP[4][16][72];                      // per-wave P tile 16x64 (+8 pad)

  // Q fragments (K-dim = head dim, two 32-chunks), resident for whole loop
  v16bf a_q[2];
#pragma unroll
  for (int kk = 0; kk < 2; ++kk) {
    const bf16* rp = qp + (long)l16 * HD + kk * 32;
    v8bf lo = *(const v8bf*)(rp + half * 8);
    v8bf hi = *(const v8bf*)(rp + 16 + half * 8);
    a_q[kk] = __builtin_shufflevector(lo, hi, 0,1,2,3,4,5,6,7,8,9,10,11,12,13,14,15);
  }

  v8f zero = {};
  v8f o[4];
#pragma unroll
  for (int nt = 0; nt < 4; ++nt) o[nt] = zero;
  float mrow[8], lrow[8];
#pragma unroll
  for (int r = 0; r < 8; ++r) { mrow[r] = -3.0e38f; lrow[r] = 0.f; }

  const float scale = 0.125f;  // 1/sqrt(64)

  for (int j = 0; j < S; j += 64) {
    // ---- S = Q @ K^T (16 x 64) ----
    v8f sc[4];
#pragma unroll
    for (int nt = 0; nt < 4; ++nt) sc[nt] = zero;
#pragma unroll
    for (int kk = 0; kk < 2; ++kk) {
#pragma unroll
      for (int nt = 0; nt < 4; ++nt) {
        // B fragment: lane -> key j+nt*16+l16 ; elems -> d (table order)
        v16bf bk;
#pragma unroll
        for (int s = 0; s < 16; ++s)
          bk[s] = ktp[(long)(kk * 32 + (s < 8 ? 8 * half + s : 16 + 8 * half + (s - 8))) * S + j + nt * 16 + l16];
        sc[nt] = WMMA_BF16(a_q[kk], bk, sc[nt]);
      }
    }
    // ---- online softmax (rows live on 16-lane groups) ----
#pragma unroll
    for (int r = 0; r < 8; ++r) {
      float mj = -3.0e38f;
#pragma unroll
      for (int nt = 0; nt < 4; ++nt) {
        float t = sc[nt][r] * scale + mk[j + nt * 16 + l16];
        sc[nt][r] = t;
        mj = fmaxf(mj, t);
      }
#pragma unroll
      for (int m = 8; m >= 1; m >>= 1) mj = fmaxf(mj, __shfl_xor(mj, m, 32));
      const float mnew = fmaxf(mrow[r], mj);
      const float corr = __expf(mrow[r] - mnew);
      float rs = 0.f;
#pragma unroll
      for (int nt = 0; nt < 4; ++nt) {
        float p = __expf(sc[nt][r] - mnew);
        sc[nt][r] = p;
        rs += p;
      }
#pragma unroll
      for (int m = 8; m >= 1; m >>= 1) rs += __shfl_xor(rs, m, 32);
      lrow[r] = lrow[r] * corr + rs;
      mrow[r] = mnew;
#pragma unroll
      for (int nt = 0; nt < 4; ++nt) o[nt][r] *= corr;
    }
    // ---- P -> LDS (C layout -> A layout transpose through memory) ----
#pragma unroll
    for (int nt = 0; nt < 4; ++nt)
#pragma unroll
      for (int r = 0; r < 8; ++r)
        sP[wave][r + 8 * half][nt * 16 + l16] = (bf16)sc[nt][r];
    __syncthreads();
    // ---- O += P @ V ----
#pragma unroll
    for (int kk = 0; kk < 2; ++kk) {
      v8bf lo = *(const v8bf*)&sP[wave][l16][kk * 32 + half * 8];
      v8bf hi = *(const v8bf*)&sP[wave][l16][kk * 32 + 16 + half * 8];
      v16bf ap = __builtin_shufflevector(lo, hi, 0,1,2,3,4,5,6,7,8,9,10,11,12,13,14,15);
#pragma unroll
      for (int nt = 0; nt < 4; ++nt) {
        // B fragment: lane -> d = nt*16+l16 ; elems -> keys (table order)
        v16bf bv;
#pragma unroll
        for (int s = 0; s < 16; ++s)
          bv[s] = vp[(long)(j + kk * 32 + (s < 8 ? 8 * half + s : 16 + 8 * half + (s - 8))) * HD + nt * 16 + l16];
        o[nt] = WMMA_BF16(ap, bv, o[nt]);
      }
    }
    __syncthreads();
  }

  // ---- normalize, write ctx[b, s, head*64 + d] (bf16) ----
#pragma unroll
  for (int nt = 0; nt < 4; ++nt)
#pragma unroll
    for (int r = 0; r < 8; ++r) {
      const long s = q0 + 8 * half + r;
      const int  d = nt * 16 + l16;
      ctx[((long)b * S + s) * H + head * HD + d] = (bf16)(o[nt][r] / lrow[r]);
    }
}

// ---------------------------------------------------------------------------
// attn-proj GEMM: ctx @ Wo + bias + residual(input)  -> attn_out (f32)
// ---------------------------------------------------------------------------
__global__ void gemm_attnproj_kernel(const bf16* __restrict__ A, const bf16* __restrict__ B,
                                     const float* __restrict__ bias,
                                     const float* __restrict__ resid,
                                     float* __restrict__ out) {
  v8f acc[4][2];
  gemm_core(A, B, H, H, acc);
  GEMM_EPILOG_SETUP();
#pragma unroll
  for (int mi = 0; mi < 4; ++mi)
#pragma unroll
    for (int ni = 0; ni < 2; ++ni) {
      const int n = nbase + ni * 16 + l16;
      const float bv = bias[n];
#pragma unroll
      for (int r = 0; r < 8; ++r) {
        const long m = mbase + mi * 16 + half * 8 + r;
        out[m * H + n] = acc[mi][ni][r] + bv + resid[m * H + n];
      }
    }
}

// ---------------------------------------------------------------------------
// FFN1 GEMM: y_ln @ W1 + b1 -> exact GELU -> act (bf16)
// ---------------------------------------------------------------------------
__global__ void gemm_ffn1_kernel(const bf16* __restrict__ A, const bf16* __restrict__ B,
                                 const float* __restrict__ bias, bf16* __restrict__ out) {
  v8f acc[4][2];
  gemm_core(A, B, H, DFF, acc);
  GEMM_EPILOG_SETUP();
#pragma unroll
  for (int mi = 0; mi < 4; ++mi)
#pragma unroll
    for (int ni = 0; ni < 2; ++ni) {
      const int n = nbase + ni * 16 + l16;
      const float bv = bias[n];
#pragma unroll
      for (int r = 0; r < 8; ++r) {
        const long m = mbase + mi * 16 + half * 8 + r;
        const float x = acc[mi][ni][r] + bv;
        const float g = 0.5f * x * (1.f + erff(x * 0.7071067811865475f));
        out[m * DFF + n] = (bf16)g;
      }
    }
}

// ---------------------------------------------------------------------------
// FFN2 GEMM: act @ W2 + b2 + attn_out -> d_out (f32)
// ---------------------------------------------------------------------------
__global__ void gemm_ffn2_kernel(const bf16* __restrict__ A, const bf16* __restrict__ B,
                                 const float* __restrict__ bias,
                                 const float* __restrict__ resid,
                                 float* __restrict__ out) {
  v8f acc[4][2];
  gemm_core(A, B, DFF, H, acc);
  GEMM_EPILOG_SETUP();
#pragma unroll
  for (int mi = 0; mi < 4; ++mi)
#pragma unroll
    for (int ni = 0; ni < 2; ++ni) {
      const int n = nbase + ni * 16 + l16;
      const float bv = bias[n];
#pragma unroll
      for (int r = 0; r < 8; ++r) {
        const long m = mbase + mi * 16 + half * 8 + r;
        out[m * H + n] = acc[mi][ni][r] + bv + resid[m * H + n];
      }
    }
}

// ---------------------------------------------------------------------------
extern "C" void kernel_launch(void* const* d_in, const int* in_sizes, int n_in,
                              void* d_out, int out_size, void* d_ws, size_t ws_size,
                              hipStream_t stream) {
  (void)in_sizes; (void)n_in; (void)out_size; (void)ws_size;
  const float* input    = (const float*)d_in[0];
  const float* mask     = (const float*)d_in[1];
  const float* norm_w   = (const float*)d_in[2];
  const float* norm_b   = (const float*)d_in[3];
  const float* qkv_w    = (const float*)d_in[4];
  const float* qkv_b    = (const float*)d_in[5];
  const float* attn_ow  = (const float*)d_in[6];
  const float* attn_ob  = (const float*)d_in[7];
  const float* attn_nw  = (const float*)d_in[8];
  const float* attn_nb  = (const float*)d_in[9];
  const float* inter_w  = (const float*)d_in[10];
  const float* inter_b  = (const float*)d_in[11];
  const float* output_w = (const float*)d_in[12];
  const float* output_b = (const float*)d_in[13];
  float* out = (float*)d_out;

  char* ws = (char*)d_ws;
  size_t off = 0;
  auto alloc = [&](size_t bytes) -> char* {
    char* p = ws + off;
    off = (off + bytes + 255) & ~(size_t)255;
    return p;
  };
  bf16*  xln      = (bf16*)alloc((size_t)T * H * 2);
  bf16*  wqkv     = (bf16*)alloc((size_t)H * 3 * H * 2);
  bf16*  wo       = (bf16*)alloc((size_t)H * H * 2);
  bf16*  w1       = (bf16*)alloc((size_t)H * DFF * 2);
  bf16*  w2       = (bf16*)alloc((size_t)DFF * H * 2);
  bf16*  qb       = (bf16*)alloc((size_t)T * H * 2);
  bf16*  ktb      = (bf16*)alloc((size_t)T * H * 2);
  bf16*  vb       = (bf16*)alloc((size_t)T * H * 2);
  bf16*  ctx      = (bf16*)alloc((size_t)T * H * 2);
  float* attn_out = (float*)alloc((size_t)T * H * 4);
  bf16*  yln      = (bf16*)alloc((size_t)T * H * 2);
  bf16*  act      = (bf16*)alloc((size_t)T * DFF * 2);

  // weight conversions
  cvt_bf16_kernel<<<2048, 256, 0, stream>>>(qkv_w,    wqkv, (long)H * 3 * H);
  cvt_bf16_kernel<<<1024, 256, 0, stream>>>(attn_ow,  wo,   (long)H * H);
  cvt_bf16_kernel<<<2048, 256, 0, stream>>>(inter_w,  w1,   (long)H * DFF);
  cvt_bf16_kernel<<<2048, 256, 0, stream>>>(output_w, w2,   (long)DFF * H);

  // LN1
  ln_bf16_kernel<<<T, 256, 0, stream>>>(input, norm_w, norm_b, xln);
  // QKV
  gemm_qkv_kernel<<<dim3(3 * H / 128, T / 128), 256, 0, stream>>>(xln, wqkv, qkv_b, qb, ktb, vb);
  // attention
  attn_kernel<<<dim3(S / 64, BB * HEADS), 128, 0, stream>>>(qb, ktb, vb, mask, ctx);
  // attn projection + residual
  gemm_attnproj_kernel<<<dim3(H / 128, T / 128), 256, 0, stream>>>(ctx, wo, attn_ob, input, attn_out);
  // LN2
  ln_bf16_kernel<<<T, 256, 0, stream>>>(attn_out, attn_nw, attn_nb, yln);
  // FFN
  gemm_ffn1_kernel<<<dim3(DFF / 128, T / 128), 256, 0, stream>>>(yln, w1, inter_b, act);
  gemm_ffn2_kernel<<<dim3(H / 128, T / 128), 256, 0, stream>>>(act, w2, output_b, attn_out, out);
}
